// TransformerSelfAttentionHead_13288628813908
// MI455X (gfx1250) — compile-verified, matrix operand order checked by reading, not run on hardware
//
#include <hip/hip_runtime.h>
#include <hip/hip_bf16.h>

// ---------- types ----------
typedef __attribute__((ext_vector_type(16))) __bf16 v16bf;
typedef __attribute__((ext_vector_type(8)))  __bf16 bf16x8;
typedef __attribute__((ext_vector_type(4)))  __bf16 bf16x4;
typedef __attribute__((ext_vector_type(8)))  float  v8f;
typedef __attribute__((ext_vector_type(4)))  int    v4i;

union Frag16 { v16bf v; bf16x8 h[2]; };

__device__ __forceinline__ __bf16 f2bf(float f) {
    union { float f; unsigned u; } x; x.f = f;
    unsigned r = x.u + 0x7fffu + ((x.u >> 16) & 1u);   // round-to-nearest-even
    union { unsigned short s; __bf16 b; } y; y.s = (unsigned short)(r >> 16);
    return y.b;
}

// ---------- async global->LDS (CDNA5 GLOBAL_LOAD_ASYNC_TO_LDS_B128) ----------
#if __has_builtin(__builtin_amdgcn_global_load_async_to_lds_b128)
#define HAVE_ASYNC_LDS 1
typedef __attribute__((address_space(1))) v4i* gv4i_p;   // pointer to int4 in global AS
typedef __attribute__((address_space(3))) v4i* lv4i_p;   // pointer to int4 in LDS AS
#endif

__device__ __forceinline__ void async_cp16(const __bf16* g, __bf16* l) {
#if defined(HAVE_ASYNC_LDS)
    __builtin_amdgcn_global_load_async_to_lds_b128(
        (gv4i_p)(unsigned long long)g,
        (lv4i_p)(unsigned)(unsigned long long)l,
        0, 0);
#else
    *(bf16x8*)l = *(const bf16x8*)g;
#endif
}

__device__ __forceinline__ void wait_async0() {
#if defined(HAVE_ASYNC_LDS)
#if __has_builtin(__builtin_amdgcn_s_wait_asynccnt)
    __builtin_amdgcn_s_wait_asynccnt(0);
#else
    asm volatile("s_wait_asynccnt 0x0" ::: "memory");
#endif
#endif
}

// ---------------------------------------------------------------------------
// Tiled WMMA GEMM:  C[M,N] = scale * A[M,K] * B^T  (+bias, relu)
//   A source: fp32 (convert->bf16 while staging) or bf16 (async DMA to LDS)
//   B source rows are the *columns* of the math-B: Bsrc[n][k] (row-major, ldb)
//   OUT_MODE 0: bf16, out[m*ldo + n]
//   OUT_MODE 1: bf16 transposed per batch-of-2048: out[(m>>11)*128*2048 + n*2048 + (m&2047)]
//   OUT_MODE 2: fp32,  out[bz*strideO + m*ldo + n]
// Block: 256 threads (8 waves). Tile 128x128, BK=64, double-buffered LDS.
// ---------------------------------------------------------------------------
template<bool A_F32, bool B_F32, bool BIAS_RELU, int OUT_MODE>
__global__ __launch_bounds__(256)
void gemm_wmma(const void* __restrict__ Ap, const void* __restrict__ Bp,
               const float* __restrict__ bias, void* __restrict__ Op,
               int M, int N, int Kd, int lda, int ldb, int ldo,
               long strideA, long strideB, long strideO, float scale)
{
    __shared__ __bf16 lA[2][128 * 72];
    __shared__ __bf16 lB[2][128 * 72];

    const int tid  = threadIdx.x;
    const int lane = tid & 31;
    const int wave = tid >> 5;
    const int bz   = blockIdx.z;
    const int gm0  = blockIdx.y * 128;
    const int gn0  = blockIdx.x * 128;

    const float*  Af = (const float*)Ap + (size_t)bz * (size_t)strideA;
    const __bf16* Ab = (const __bf16*)Ap + (size_t)bz * (size_t)strideA;
    const float*  Bf = (const float*)Bp + (size_t)bz * (size_t)strideB;
    const __bf16* Bb = (const __bf16*)Bp + (size_t)bz * (size_t)strideB;

    v8f acc[8];
#pragma unroll
    for (int i = 0; i < 8; ++i)
#pragma unroll
        for (int j = 0; j < 8; ++j) acc[i][j] = 0.0f;

    float4 ra[8];   // next-tile register stage (fp32 paths only)
    float4 rb[8];

    // ---- staging helpers (inlined; template bools prune dead paths) ----
    // fp32 path: load (early) / convert+store (late)
    auto loadA_f32 = [&](int k0) {
#pragma unroll
        for (int i = 0; i < 8; ++i) {
            int c = tid + i * 256, row = c >> 4, kq = (c & 15) * 4;
            ra[i] = *(const float4*)&Af[(long)(gm0 + row) * lda + k0 + kq];
        }
    };
    auto storeA_f32 = [&](int buf) {
#pragma unroll
        for (int i = 0; i < 8; ++i) {
            int c = tid + i * 256, row = c >> 4, kq = (c & 15) * 4;
            bf16x4 o; o[0] = f2bf(ra[i].x); o[1] = f2bf(ra[i].y);
            o[2] = f2bf(ra[i].z); o[3] = f2bf(ra[i].w);
            *(bf16x4*)&lA[buf][row * 72 + kq] = o;
        }
    };
    auto loadB_f32 = [&](int k0) {
#pragma unroll
        for (int i = 0; i < 8; ++i) {
            int c = tid + i * 256, row = c >> 4, kq = (c & 15) * 4;
            rb[i] = *(const float4*)&Bf[(long)(gn0 + row) * ldb + k0 + kq];
        }
    };
    auto storeB_f32 = [&](int buf) {
#pragma unroll
        for (int i = 0; i < 8; ++i) {
            int c = tid + i * 256, row = c >> 4, kq = (c & 15) * 4;
            bf16x4 o; o[0] = f2bf(rb[i].x); o[1] = f2bf(rb[i].y);
            o[2] = f2bf(rb[i].z); o[3] = f2bf(rb[i].w);
            *(bf16x4*)&lB[buf][row * 72 + kq] = o;
        }
    };
    // bf16 path: async DMA straight into LDS
    auto asyncA_bf16 = [&](int buf, int k0) {
#pragma unroll
        for (int i = 0; i < 4; ++i) {
            int c = tid + i * 256, row = c >> 3, kq = (c & 7) * 8;
            async_cp16(&Ab[(long)(gm0 + row) * lda + k0 + kq], &lA[buf][row * 72 + kq]);
        }
    };
    auto asyncB_bf16 = [&](int buf, int k0) {
#pragma unroll
        for (int i = 0; i < 4; ++i) {
            int c = tid + i * 256, row = c >> 3, kq = (c & 7) * 8;
            async_cp16(&Bb[(long)(gn0 + row) * ldb + k0 + kq], &lB[buf][row * 72 + kq]);
        }
    };

    // ---- prologue: stage tile 0 into buffer 0 ----
    if (A_F32) { loadA_f32(0); } else { asyncA_bf16(0, 0); }
    if (B_F32) { loadB_f32(0); } else { asyncB_bf16(0, 0); }
    if (A_F32) storeA_f32(0);
    if (B_F32) storeB_f32(0);
    wait_async0();
    __syncthreads();

    int buf = 0;
    for (int k0 = 0; k0 < Kd; k0 += 64, buf ^= 1) {
        const bool more = (k0 + 64) < Kd;
        // issue next tile's loads before compute (async writes other buffer)
        if (more) {
            if (A_F32) loadA_f32(k0 + 64);   else asyncA_bf16(buf ^ 1, k0 + 64);
            if (B_F32) loadB_f32(k0 + 64);   else asyncB_bf16(buf ^ 1, k0 + 64);
            __builtin_prefetch(A_F32 ? (const void*)&Af[(long)(gm0 + (tid >> 1)) * lda + k0 + 64]
                                     : (const void*)&Ab[(long)(gm0 + (tid >> 1)) * lda + k0 + 64], 0, 1);
        }

        // -------- compute: 2 K-steps of 32, 8 N-tiles, B-frags pipelined --------
#pragma unroll
        for (int kk = 0; kk < 64; kk += 32) {
            Frag16 a;
            const int arow = wave * 16 + (lane & 15);
            const int akb  = kk + ((lane >> 4) << 3);     // lanes 16-31 take K+8
            const __bf16* As = &lA[buf][arow * 72 + akb];
            a.h[0] = *(const bf16x8*)As;
            a.h[1] = *(const bf16x8*)(As + 16);
            Frag16 bc;
            {
                const __bf16* Bs = &lB[buf][(lane & 15) * 72 + akb];
                bc.h[0] = *(const bf16x8*)Bs;
                bc.h[1] = *(const bf16x8*)(Bs + 16);
            }
#pragma unroll
            for (int nt = 0; nt < 8; ++nt) {
                Frag16 bn;
                if (nt < 7) {
                    const __bf16* Bs = &lB[buf][((nt + 1) * 16 + (lane & 15)) * 72 + akb];
                    bn.h[0] = *(const bf16x8*)Bs;
                    bn.h[1] = *(const bf16x8*)(Bs + 16);
                }
                acc[nt] = __builtin_amdgcn_wmma_f32_16x16x32_bf16(
                    false, a.v, false, bc.v, (short)0, acc[nt], false, false);
                if (nt < 7) bc = bn;
            }
        }

        // finish staging the next tile, then one barrier per K-step
        if (more) {
            if (A_F32) storeA_f32(buf ^ 1);
            if (B_F32) storeB_f32(buf ^ 1);
        }
        wait_async0();
        __syncthreads();
    }

    // -------- epilogue --------
    const int mloc  = wave * 16 + ((lane >> 4) << 3);   // + r
    const int nbase = lane & 15;
#pragma unroll
    for (int nt = 0; nt < 8; ++nt) {
        const int gn = gn0 + nt * 16 + nbase;
        float bi = 0.0f;
        if (BIAS_RELU) bi = bias[gn];
#pragma unroll
        for (int r = 0; r < 8; ++r) {
            const int gm = gm0 + mloc + r;
            float v = acc[nt][r] * scale;
            if (BIAS_RELU) { v += bi; v = v > 0.0f ? v : 0.0f; }
            if (OUT_MODE == 0) {
                ((__bf16*)Op)[(long)gm * ldo + gn] = f2bf(v);
            } else if (OUT_MODE == 1) {
                const int bb = gm >> 11, s = gm & 2047;
                ((__bf16*)Op)[(long)bb * (128L * 2048L) + (long)gn * 2048L + s] = f2bf(v);
            } else {
                ((float*)Op)[(long)bz * strideO + (long)gm * ldo + gn] = v;
            }
        }
    }
}

// ---------------------------------------------------------------------------
// Softmax over axis=1 (query axis): one thread per column (b,k); coalesced.
// ---------------------------------------------------------------------------
__global__ __launch_bounds__(128)
void col_softmax(float* __restrict__ X)
{
    const int b = blockIdx.y;
    const int k = blockIdx.x * 128 + threadIdx.x;
    float* p = X + (long)b * 2048L * 2048L + k;
    float mx = -1e30f;
    for (int q = 0; q < 2048; ++q) mx = fmaxf(mx, p[(long)q * 2048]);
    float sum = 0.0f;
    for (int q = 0; q < 2048; ++q) sum += __expf(p[(long)q * 2048] - mx);
    const float inv = 1.0f / sum;
    for (int q = 0; q < 2048; ++q) p[(long)q * 2048] = __expf(p[(long)q * 2048] - mx) * inv;
}

// ---------------------------------------------------------------------------
// Softmax over axis=2 (key axis): one 256-thread block per row, in place.
// ---------------------------------------------------------------------------
__global__ __launch_bounds__(256)
void row_softmax(float* __restrict__ X)
{
    __shared__ float red[256];
    const long row = blockIdx.x;                 // 0..16383
    float* p = X + row * 2048L;
    const int tid = threadIdx.x;

    float mx = -1e30f;
    for (int i = tid; i < 2048; i += 256) mx = fmaxf(mx, p[i]);
    red[tid] = mx; __syncthreads();
    for (int s = 128; s > 0; s >>= 1) { if (tid < s) red[tid] = fmaxf(red[tid], red[tid + s]); __syncthreads(); }
    mx = red[0]; __syncthreads();

    float sum = 0.0f;
    for (int i = tid; i < 2048; i += 256) sum += __expf(p[i] - mx);
    red[tid] = sum; __syncthreads();
    for (int s = 128; s > 0; s >>= 1) { if (tid < s) red[tid] += red[tid + s]; __syncthreads(); }
    const float inv = 1.0f / red[0];

    for (int i = tid; i < 2048; i += 256) p[i] = __expf(p[i] - mx) * inv;
}

// ---------------------------------------------------------------------------
extern "C" void kernel_launch(void* const* d_in, const int* in_sizes, int n_in,
                              void* d_out, int out_size, void* d_ws, size_t ws_size,
                              hipStream_t stream)
{
    (void)in_sizes; (void)n_in; (void)out_size; (void)ws_size;

    const float* X  = (const float*)d_in[0];   // [8,2048,1024]
    const float* WQ = (const float*)d_in[1];   // [128,1024]
    const float* bQ = (const float*)d_in[2];
    const float* WK = (const float*)d_in[3];
    const float* bK = (const float*)d_in[4];
    const float* WV = (const float*)d_in[5];
    const float* bV = (const float*)d_in[6];
    float* Z = (float*)d_out;                  // [8,2048,128] fp32

    char* ws = (char*)d_ws;
    __bf16* Qb = (__bf16*)(ws);                         //  4 MB: [16384,128] bf16
    __bf16* Kb = (__bf16*)(ws + ((size_t)4  << 20));    //  4 MB: [16384,128] bf16
    __bf16* Vt = (__bf16*)(ws + ((size_t)8  << 20));    //  4 MB: [8,128,2048] bf16 (transposed)
    float*  P  = (float*)(ws + ((size_t)12 << 20));     // 128 MB: [8,2048,2048] fp32

    const dim3 blk(256);

    // 1) QKV projections: [16384,1024] x [1024,128] + bias, relu  -> bf16
    gemm_wmma<true, true, true, 0><<<dim3(1, 128, 1), blk, 0, stream>>>(
        X, WQ, bQ, Qb, 16384, 128, 1024, 1024, 1024, 128, 0, 0, 0, 1.0f);
    gemm_wmma<true, true, true, 0><<<dim3(1, 128, 1), blk, 0, stream>>>(
        X, WK, bK, Kb, 16384, 128, 1024, 1024, 1024, 128, 0, 0, 0, 1.0f);
    gemm_wmma<true, true, true, 1><<<dim3(1, 128, 1), blk, 0, stream>>>(
        X, WV, bV, Vt, 16384, 128, 1024, 1024, 1024, 2048, 0, 0, 0, 1.0f);

    // 2) scores: per batch  Q[2048,128] x K^T -> P fp32, scaled by 1/sqrt(1024)=1/32
    gemm_wmma<false, false, false, 2><<<dim3(16, 16, 8), blk, 0, stream>>>(
        Qb, Kb, nullptr, P, 2048, 2048, 128, 128, 128, 2048,
        2048L * 128L, 2048L * 128L, 2048L * 2048L, 0.03125f);

    // 3) softmax over query axis (columns), then 4) over key axis (rows), in place
    col_softmax<<<dim3(16, 8), 128, 0, stream>>>(P);
    row_softmax<<<16384, 256, 0, stream>>>(P);

    // 5) Z = P x V : per batch [2048,2048] x [2048,128] -> fp32 out
    gemm_wmma<true, false, false, 2><<<dim3(1, 16, 8), blk, 0, stream>>>(
        P, Vt, nullptr, Z, 2048, 128, 2048, 2048, 2048, 128,
        2048L * 2048L, 128L * 2048L, 2048L * 128L, 1.0f);
}
